// HGGN_56014963474545
// MI455X (gfx1250) — matrix operand order repaired and checked by reading, not compiled, not run on hardware
//
#include <hip/hip_runtime.h>
#include <hip/hip_bf16.h>
#include <math.h>

// ---------------- problem constants (from reference) ----------------
constexpr int NN      = 6016;       // N_NODE
constexpr int NF      = 128;
constexpr int R       = 64;
constexpr int NEDGE   = 300000;     // E
constexpr int ETOT    = NEDGE + NN; // edges + self loops
constexpr int NUM_RNA = 4000;
constexpr int P       = 500000;

typedef __attribute__((ext_vector_type(16))) _Float16 v16h;
typedef __attribute__((ext_vector_type(8)))  float    v8f;

union AFrag { v16h h; uint4 q[2]; };

// ---------------- WMMA fragment loaders (CDNA5 16-bit layouts) ----------------
// A (16x32 f16): lane<16 -> row M=lane, K = {0..7, 16..23}; lane>=16 -> M=lane-16, K = {8..15, 24..31}
__device__ __forceinline__ v16h load_a_frag(const _Float16* __restrict__ act, int strideH,
                                            int lane, int kTile) {
    const int row = lane & 15, hi = lane >> 4;
    const int k0 = kTile * 32 + hi * 8;
    AFrag a;
    a.q[0] = *(const uint4*)(act + (size_t)row * strideH + k0);
    a.q[1] = *(const uint4*)(act + (size_t)row * strideH + k0 + 16);
    return a.h;
}

// B fragments pre-packed in fragment-major order: frag (kt,nt) -> 32 lanes x 16 contiguous halves
__device__ __forceinline__ v16h load_b_packed(const _Float16* __restrict__ Wp,
                                              int nTiles, int lane, int kt, int nt) {
    const _Float16* p = Wp + (((size_t)(kt * nTiles + nt) * 32) + lane) * 16;
    AFrag b;
    b.q[0] = *(const uint4*)p;
    b.q[1] = *(const uint4*)(p + 8);
    return b.h;
}

// ---------------- utility kernels ----------------
__global__ void k_f32_to_f16(const float* __restrict__ s, _Float16* __restrict__ d, int n) {
    int t = blockIdx.x * blockDim.x + threadIdx.x;
    if (t < n) d[t] = (_Float16)s[t];
}

// pack a (realK x realN) f32 weight, zero-padded to (K x N), into WMMA B-fragment order:
// out[((kt*nTiles+nt)*32 + lane)*16 + j] = W[kt*32 + hi*8 + (j&7) + (j>>3)*16][nt*16 + row]
__global__ void k_pack_b(const float* __restrict__ W, _Float16* __restrict__ out,
                         int realK, int realN, int K, int N) {
    int t = blockIdx.x * blockDim.x + threadIdx.x;
    if (t >= K * N) return;
    int nTiles = N >> 4;
    int frag = t >> 9;          // 512 halves per fragment
    int rem  = t & 511;
    int lane = rem >> 4;
    int j    = rem & 15;
    int kt = frag / nTiles, nt = frag % nTiles;
    int row = lane & 15, hi = lane >> 4;
    int k = kt * 32 + hi * 8 + (j & 7) + (j >> 3) * 16;
    int n = nt * 16 + row;
    float v = (k < realK && n < realN) ? W[(size_t)k * realN + n] : 0.f;
    out[t] = (_Float16)v;
}

__global__ void k_init(int* __restrict__ menc, float* __restrict__ denom, float* __restrict__ accum) {
    int t = blockIdx.x * blockDim.x + threadIdx.x;
    if (t < NN) { menc[t] = (int)0x807FFFFF; denom[t] = 0.f; } // enc(-inf)
    if (t < NN * R) accum[t] = 0.f;
}

// ---------------- node GEMM: H(NNx64) = A(NNxK, f16) @ W(Kx64 packed), f32 out ----------------
template <int K>
__global__ __launch_bounds__(64) void k_gemm_nodes(const _Float16* __restrict__ A,
                                                   const _Float16* __restrict__ Wp,
                                                   float* __restrict__ H) {
    constexpr int KT = K / 32;
    const int tid = threadIdx.x;
    const int wave = tid >> 5, lane = tid & 31;
    const int tile = blockIdx.x * 2 + wave;      // 376 tiles exactly, grid = 188
    const int row0 = tile * 16;
    const int row = lane & 15, hi = lane >> 4;
    const _Float16* Arow = A + (size_t)(row0 + row) * K;

    v16h a[KT];
#pragma unroll
    for (int kt = 0; kt < KT; ++kt) {
        AFrag af;
        const int k0 = kt * 32 + hi * 8;
        af.q[0] = *(const uint4*)(Arow + k0);
        af.q[1] = *(const uint4*)(Arow + k0 + 16);
        a[kt] = af.h;
    }
#pragma unroll
    for (int nt = 0; nt < 4; ++nt) {
        v8f acc = {};
#pragma unroll
        for (int kt = 0; kt < KT; ++kt) {
            v16h b = load_b_packed(Wp, 4, lane, kt, nt);
            acc = __builtin_amdgcn_wmma_f32_16x16x32_f16(false, a[kt], false, b,
                                                         (short)0, acc, false, false);
        }
#pragma unroll
        for (int r = 0; r < 8; ++r)
            H[(size_t)(row0 + r + hi * 8) * 64 + nt * 16 + row] = acc[r];
    }
}

// ---------------- GAT attention pipeline ----------------
__global__ void k_scores(const float* __restrict__ H, const float* __restrict__ as,
                         const float* __restrict__ ad, float* __restrict__ ss,
                         float* __restrict__ sd) {
    int n = blockIdx.x * blockDim.x + threadIdx.x;
    if (n >= NN) return;
    float a = 0.f, b = 0.f;
    for (int k = 0; k < 64; ++k) {
        float h = H[(size_t)n * 64 + k];
        a += h * as[k]; b += h * ad[k];
    }
    ss[n] = a; sd[n] = b;
}

__device__ __forceinline__ int enc_f32(float v) {
    int i = __float_as_int(v);
    return i < 0 ? (i ^ 0x7FFFFFFF) : i;
}

__global__ void k_edge_e(const int* __restrict__ adj, const float* __restrict__ ss,
                         const float* __restrict__ sd, float* __restrict__ ebuf,
                         int* __restrict__ menc) {
    int e = blockIdx.x * blockDim.x + threadIdx.x;
    if (e >= ETOT) return;
    int s, d;
    if (e < NEDGE) { s = adj[e]; d = adj[NEDGE + e]; } else { s = d = e - NEDGE; }
    float v = ss[s] + sd[d];
    v = v > 0.f ? v : 0.2f * v;              // leaky_relu(0.2)
    ebuf[e] = v;
    atomicMax(menc + d, enc_f32(v));
}

__global__ void k_edge_exp(const int* __restrict__ adj, float* __restrict__ ebuf,
                           const int* __restrict__ menc, float* __restrict__ denom) {
    int e = blockIdx.x * blockDim.x + threadIdx.x;
    if (e >= ETOT) return;
    int d = e < NEDGE ? adj[NEDGE + e] : e - NEDGE;
    int me = menc[d]; me = me < 0 ? (me ^ 0x7FFFFFFF) : me;
    float ex = expf(ebuf[e] - __int_as_float(me));
    ebuf[e] = ex;
    atomicAdd(denom + d, ex);
}

// one thread per (edge, 4 features)
__global__ void k_aggr(const int* __restrict__ adj, const float* __restrict__ ebuf,
                       const float* __restrict__ H, float* __restrict__ accum) {
    int t = blockIdx.x * blockDim.x + threadIdx.x;
    int e = t >> 4;
    if (e >= ETOT) return;
    int f = (t & 15) * 4;
    int s, d;
    if (e < NEDGE) { s = adj[e]; d = adj[NEDGE + e]; } else { s = d = e - NEDGE; }
    float w = ebuf[e];
    float4 hv = *(const float4*)(H + (size_t)s * 64 + f);
    float* ap = accum + (size_t)d * 64 + f;
    atomicAdd(ap + 0, w * hv.x);
    atomicAdd(ap + 1, w * hv.y);
    atomicAdd(ap + 2, w * hv.z);
    atomicAdd(ap + 3, w * hv.w);
}

__global__ void k_finalize(const float* __restrict__ accum, const float* __restrict__ denom,
                           const float* __restrict__ gb, _Float16* __restrict__ hh) {
    int t = blockIdx.x * blockDim.x + threadIdx.x;
    if (t >= NN * R) return;
    int n = t >> 6, f = t & 63;
    hh[t] = (_Float16)(accum[t] / denom[n] + gb[f]);
}

// ---------------- pair MLP (WMMA), 16 pairs per wave ----------------
template <int K, int N>
__device__ __forceinline__ void mlp_layer(const _Float16* __restrict__ in, int inStride,
                                          const _Float16* __restrict__ Wp,
                                          const float* __restrict__ bias,
                                          _Float16* __restrict__ out, int outStride, int lane) {
    constexpr int KT = K / 32, NT = N / 16;
    const int row = lane & 15, hi = lane >> 4;
    v16h a[KT];
#pragma unroll
    for (int kt = 0; kt < KT; ++kt) a[kt] = load_a_frag(in, inStride, lane, kt);
#pragma unroll
    for (int nt = 0; nt < NT; ++nt) {
        v8f acc = {};
#pragma unroll
        for (int kt = 0; kt < KT; ++kt) {
            v16h b = load_b_packed(Wp, NT, lane, kt, nt);
            acc = __builtin_amdgcn_wmma_f32_16x16x32_f16(false, a[kt], false, b,
                                                         (short)0, acc, false, false);
        }
        float bv = bias[nt * 16 + row];
#pragma unroll
        for (int r = 0; r < 8; ++r) {
            float v = acc[r] + bv;
            v = v > 0.f ? v : 0.f;                       // ReLU
            out[(size_t)(r + hi * 8) * outStride + nt * 16 + row] = (_Float16)v;
        }
    }
}

__global__ __launch_bounds__(64) void k_mlp(const _Float16* __restrict__ hh,
                                            const int* __restrict__ coo,
                                            const _Float16* __restrict__ w1p, const _Float16* __restrict__ w2p,
                                            const _Float16* __restrict__ w3p, const _Float16* __restrict__ w4p,
                                            const _Float16* __restrict__ w5p,
                                            const float* __restrict__ b1, const float* __restrict__ b2,
                                            const float* __restrict__ b3, const float* __restrict__ b4,
                                            const float* __restrict__ b5,
                                            const float* __restrict__ w6, const float* __restrict__ b6,
                                            float* __restrict__ out) {
    __shared__ __align__(16) _Float16 sA[2][16 * 128];
    __shared__ __align__(16) _Float16 sB[2][16 * 64];
    __shared__ float sF[2][16 * 8];

    const int tid = threadIdx.x;
    const int wave = tid >> 5, lane = tid & 31;
    const int p0 = (blockIdx.x * 2 + wave) * 16;   // grid = P/16/2 exactly -> no guards

    // gather z = [x1[coo0] , x2[coo1]] into per-wave LDS tile (16 x 128 f16)
    {
        const int m = lane >> 1, half = lane & 1;
        const int pr = p0 + m;
        const int node = half ? (NUM_RNA + coo[2 * pr + 1]) : coo[2 * pr];
        const uint4* s = (const uint4*)(hh + (size_t)node * 64);
        uint4* d = (uint4*)(sA[wave] + m * 128 + half * 64);
#pragma unroll
        for (int i = 0; i < 8; ++i) d[i] = s[i];
    }
    __syncthreads();

    mlp_layer<128, 64>(sA[wave], 128, w1p, b1, sB[wave], 64, lane);   // 16 WMMA
    __syncthreads();
    mlp_layer<64, 32>(sB[wave], 64, w2p, b2, sA[wave], 128, lane);    // 4 WMMA
    __syncthreads();
    mlp_layer<32, 32>(sA[wave], 128, w3p, b3, sB[wave], 64, lane);    // 2 WMMA
    __syncthreads();
    mlp_layer<32, 16>(sB[wave], 64, w4p, b4, sA[wave], 128, lane);    // 1 WMMA
    // zero-pad K cols 16..31 for layer-5's 16x16x32 tile
    {
        const int row = lane & 15, hi = lane >> 4;
#pragma unroll
        for (int r = 0; r < 8; ++r)
            sA[wave][(size_t)(r + hi * 8) * 128 + 16 + row] = (_Float16)0.f;
    }
    __syncthreads();
    // layer 5: K=16 (padded to 32), N=8 (padded to 16), f32 result to LDS
    {
        const int row = lane & 15, hi = lane >> 4;
        v8f acc = {};
        v16h a = load_a_frag(sA[wave], 128, lane, 0);
        v16h b = load_b_packed(w5p, 1, lane, 0, 0);
        acc = __builtin_amdgcn_wmma_f32_16x16x32_f16(false, a, false, b,
                                                     (short)0, acc, false, false); // 1 WMMA
        if (row < 8) {
            float bv = b5[row];
#pragma unroll
            for (int r = 0; r < 8; ++r) {
                float v = acc[r] + bv;
                v = v > 0.f ? v : 0.f;
                sF[wave][(r + hi * 8) * 8 + row] = v;
            }
        }
    }
    __syncthreads();
    // layer 6: 8 -> 1, sigmoid
    if (lane < 16) {
        float s = b6[0];
#pragma unroll
        for (int k = 0; k < 8; ++k) s += sF[wave][lane * 8 + k] * w6[k];
        out[p0 + lane] = 1.f / (1.f + expf(-s));
    }
}

// ---------------- host launcher ----------------
extern "C" void kernel_launch(void* const* d_in, const int* in_sizes, int n_in,
                              void* d_out, int out_size, void* d_ws, size_t ws_size,
                              hipStream_t stream) {
    (void)in_sizes; (void)n_in; (void)out_size; (void)ws_size;
    const float* x    = (const float*)d_in[0];
    const int*   adj  = (const int*)d_in[1];
    const int*   coo  = (const int*)d_in[2];
    const float* gw0  = (const float*)d_in[3];
    const float* gas0 = (const float*)d_in[4];
    const float* gad0 = (const float*)d_in[5];
    const float* gb0  = (const float*)d_in[6];
    const float* gw1  = (const float*)d_in[7];
    const float* gas1 = (const float*)d_in[8];
    const float* gad1 = (const float*)d_in[9];
    const float* gb1  = (const float*)d_in[10];
    const float* w1 = (const float*)d_in[11]; const float* b1 = (const float*)d_in[12];
    const float* w2 = (const float*)d_in[13]; const float* b2 = (const float*)d_in[14];
    const float* w3 = (const float*)d_in[15]; const float* b3 = (const float*)d_in[16];
    const float* w4 = (const float*)d_in[17]; const float* b4 = (const float*)d_in[18];
    const float* w5 = (const float*)d_in[19]; const float* b5 = (const float*)d_in[20];
    const float* w6 = (const float*)d_in[21]; const float* b6 = (const float*)d_in[22];
    float* out = (float*)d_out;

    // workspace carve-out (256B aligned), ~6.8 MB total
    char* base = (char*)d_ws; size_t off = 0;
    auto alloc = [&](size_t bytes) -> void* {
        void* p = base + off;
        off = (off + bytes + 255) & ~(size_t)255;
        return p;
    };
    _Float16* xh   = (_Float16*)alloc((size_t)NN * NF * 2);
    _Float16* gw0p = (_Float16*)alloc((size_t)NF * 64 * 2);
    _Float16* gw1p = (_Float16*)alloc((size_t)64 * 64 * 2);
    _Float16* w1p  = (_Float16*)alloc((size_t)128 * 64 * 2);
    _Float16* w2p  = (_Float16*)alloc((size_t)64 * 32 * 2);
    _Float16* w3p  = (_Float16*)alloc((size_t)32 * 32 * 2);
    _Float16* w4p  = (_Float16*)alloc((size_t)32 * 16 * 2);
    _Float16* w5p  = (_Float16*)alloc((size_t)32 * 16 * 2);
    float*    H    = (float*)alloc((size_t)NN * 64 * 4);
    float*    ss   = (float*)alloc((size_t)NN * 4);
    float*    sd   = (float*)alloc((size_t)NN * 4);
    int*      menc = (int*)alloc((size_t)NN * 4);
    float*    den  = (float*)alloc((size_t)NN * 4);
    float*    acc  = (float*)alloc((size_t)NN * 64 * 4);
    float*    ebuf = (float*)alloc((size_t)ETOT * 4);
    _Float16* hh   = (_Float16*)alloc((size_t)NN * 64 * 2);

    auto cdiv = [](int a, int b) { return (a + b - 1) / b; };

    // A-side f32 -> f16 conversion
    k_f32_to_f16<<<cdiv(NN * NF, 256), 256, 0, stream>>>(x, xh, NN * NF);
    // B-side weights: convert + pack into WMMA fragment order (zero-padded where needed)
    k_pack_b<<<cdiv(128 * 64, 256), 256, 0, stream>>>(gw0, gw0p, 128, 64, 128, 64);
    k_pack_b<<<cdiv(64 * 64, 256), 256, 0, stream>>>(gw1, gw1p, 64, 64, 64, 64);
    k_pack_b<<<cdiv(128 * 64, 256), 256, 0, stream>>>(w1, w1p, 128, 64, 128, 64);
    k_pack_b<<<cdiv(64 * 32, 256), 256, 0, stream>>>(w2, w2p, 64, 32, 64, 32);
    k_pack_b<<<cdiv(32 * 32, 256), 256, 0, stream>>>(w3, w3p, 32, 32, 32, 32);
    k_pack_b<<<cdiv(32 * 16, 256), 256, 0, stream>>>(w4, w4p, 32, 16, 32, 16);
    k_pack_b<<<cdiv(32 * 16, 256), 256, 0, stream>>>(w5, w5p, 16, 8, 32, 16);

    const int initBlocks = cdiv(NN * 64, 256);
    const int edgeBlocks = cdiv(ETOT, 256);
    const int aggrBlocks = cdiv(ETOT * 16, 256);

    // ---- GAT layer 0 ----
    k_init<<<initBlocks, 256, 0, stream>>>(menc, den, acc);
    k_gemm_nodes<128><<<NN / 32, 64, 0, stream>>>(xh, gw0p, H);          // 188 blocks
    k_scores<<<cdiv(NN, 256), 256, 0, stream>>>(H, gas0, gad0, ss, sd);
    k_edge_e<<<edgeBlocks, 256, 0, stream>>>(adj, ss, sd, ebuf, menc);
    k_edge_exp<<<edgeBlocks, 256, 0, stream>>>(adj, ebuf, menc, den);
    k_aggr<<<aggrBlocks, 256, 0, stream>>>(adj, ebuf, H, acc);
    k_finalize<<<initBlocks, 256, 0, stream>>>(acc, den, gb0, hh);

    // ---- GAT layer 1 ----
    k_init<<<initBlocks, 256, 0, stream>>>(menc, den, acc);
    k_gemm_nodes<64><<<NN / 32, 64, 0, stream>>>(hh, gw1p, H);
    k_scores<<<cdiv(NN, 256), 256, 0, stream>>>(H, gas1, gad1, ss, sd);
    k_edge_e<<<edgeBlocks, 256, 0, stream>>>(adj, ss, sd, ebuf, menc);
    k_edge_exp<<<edgeBlocks, 256, 0, stream>>>(adj, ebuf, menc, den);
    k_aggr<<<aggrBlocks, 256, 0, stream>>>(adj, ebuf, H, acc);
    k_finalize<<<initBlocks, 256, 0, stream>>>(acc, den, gb1, hh);

    // ---- pair MLP, 16 pairs per wave, 2 waves per block ----
    k_mlp<<<P / 32, 64, 0, stream>>>(hh, coo, w1p, w2p, w3p, w4p, w5p,
                                     b1, b2, b3, b4, b5, w6, b6, out);
}
